// Block_77309411620
// MI455X (gfx1250) — compile-verified, matrix operand order checked by reading, not run on hardware
//
#include <hip/hip_runtime.h>
#include <hip/hip_bf16.h>
#include <math.h>

// ---------------- problem constants ----------------
constexpr int BB  = 2;
constexpr int NN  = 300;
constexpr int LL  = 16384;
constexpr int CC  = 256;
constexpr int HH  = 8;
constexpr int DH  = 32;
constexpr int HID = 1024;
constexpr int NT  = 19;            // ceil(300/16) n-tiles
constexpr int RT  = 38;            // ceil(600/16) row-tiles over B*N
constexpr int LC  = 256;           // l-chunk per workgroup in attention
constexpr int NCHUNK = LL / LC;    // 64
#define SCALE 0.17677669529663687f // 32^-0.5

// ---------------- WMMA types / helpers ----------------
typedef __bf16 v16bf __attribute__((ext_vector_type(16)));
typedef float  v8f   __attribute__((ext_vector_type(8)));

struct B16x8 { unsigned int d[4]; };            // 16 bytes of 8 bf16
union V16BF { v16bf v; __bf16 h[16]; B16x8 q[2]; };

__device__ inline v8f zero8() {
  v8f z = {0.f,0.f,0.f,0.f,0.f,0.f,0.f,0.f};
  return z;
}

__device__ inline v8f wmma_bf16(v16bf a, v16bf b, v8f c) {
  return __builtin_amdgcn_wmma_f32_16x16x32_bf16(
      /*neg_a=*/false, a, /*neg_b=*/false, b,
      /*c_mod=*/(short)0, c, /*reuse_a=*/false, /*reuse_b=*/false);
}

// Build 16x32 bf16 A/B operand for lane: halves[0..7]=K[ko..ko+7],
// halves[8..15]=K[16+ko..16+ko+7]. p0/p1 are 16B-aligned bf16 pointers.
__device__ inline v16bf load_ab_bf16(const __bf16* p0, const __bf16* p1) {
  V16BF u;
  u.q[0] = *(const B16x8*)p0;
  u.q[1] = *(const B16x8*)p1;
  return u.v;
}

__device__ inline v16bf make_ab_from_f32(const float* p0, const float* p1) {
  V16BF u;
#pragma unroll
  for (int i = 0; i < 8; ++i) u.h[i]     = (__bf16)p0[i];
#pragma unroll
  for (int i = 0; i < 8; ++i) u.h[8 + i] = (__bf16)p1[i];
  return u.v;
}

// ---------------- weight transpose + fp32->bf16 ----------------
__global__ void tconv_kernel(const float* __restrict__ in, __bf16* __restrict__ out,
                             int rows, int cols) {
  int i = blockIdx.x * blockDim.x + threadIdx.x;
  if (i < rows * cols) {
    int r = i / cols, c = i % cols;
    out[(size_t)c * rows + r] = (__bf16)in[i];  // out is [cols][rows] (out-ch major)
  }
}

// ---------------- q/k/v projections ----------------
// tiles: [0,38) -> q rows (600), [38,2086) -> k rows (32768), [2086,4134) -> v
__global__ void proj_kernel(const float* __restrict__ query, const float* __restrict__ key,
                            const float* __restrict__ value,
                            const __bf16* __restrict__ WqT, const __bf16* __restrict__ WkT,
                            const __bf16* __restrict__ WvT,
                            __bf16* __restrict__ qp, __bf16* __restrict__ kp,
                            __bf16* __restrict__ vT) {
  const int tile = blockIdx.x;
  const int wave = threadIdx.x >> 5;
  const int lane = threadIdx.x & 31;
  const int m  = lane & 15;
  const int ko = (lane >> 4) << 3;

  const float* src; const __bf16* WT; int nrows, mode, t;
  if (tile < 38)        { src = query; WT = WqT; nrows = BB*NN; mode = 0; t = tile; }
  else if (tile < 2086) { src = key;   WT = WkT; nrows = BB*LL; mode = 1; t = tile - 38; }
  else                  { src = value; WT = WvT; nrows = BB*LL; mode = 2; t = tile - 2086; }

  const int r0 = t * 16;
  int rr = r0 + m; if (rr >= nrows) rr = nrows - 1;
  const float* arow = src + (size_t)rr * CC;

  v8f acc[4];
#pragma unroll
  for (int s = 0; s < 4; ++s) acc[s] = zero8();

  for (int kk = 0; kk < CC; kk += 32) {
    v16bf A = make_ab_from_f32(arow + kk + ko, arow + kk + 16 + ko);
#pragma unroll
    for (int s = 0; s < 4; ++s) {
      int ncol = wave * 64 + s * 16 + m;                 // this lane's output column
      const __bf16* wrow = WT + (size_t)ncol * CC + kk;
      v16bf Bv = load_ab_bf16(wrow + ko, wrow + 16 + ko);
      acc[s] = wmma_bf16(A, Bv, acc[s]);
    }
  }

#pragma unroll
  for (int s = 0; s < 4; ++s) {
    int ncol = wave * 64 + s * 16 + m;
    int h = ncol >> 5, dh = ncol & 31;
#pragma unroll
    for (int i = 0; i < 8; ++i) {
      int mr = i + ((lane >> 4) << 3);
      int r = r0 + mr;
      if (r >= nrows) continue;
      __bf16 vv = (__bf16)acc[s][i];
      if (mode == 0) {
        int b = r / NN, n = r % NN;
        qp[(((size_t)(b*HH + h) * NN + n) << 5) + dh] = vv;
      } else if (mode == 1) {
        int b = r >> 14, l = r & (LL - 1);
        kp[(((size_t)(b*HH + h) * LL + l) << 5) + dh] = vv;
      } else {
        int b = r >> 14, l = r & (LL - 1);
        vT[((size_t)(b*HH + h) * DH + dh) * LL + l] = vv;
      }
    }
  }
}

// ---------------- flash-attention split-K + gate-MLP mask ----------------
__global__ void attn_kernel(const __bf16* __restrict__ qp, const __bf16* __restrict__ kp,
                            const __bf16* __restrict__ vT,
                            const unsigned char* __restrict__ kpm,
                            const float* __restrict__ Wg1, const float* __restrict__ bg1,
                            const float* __restrict__ Wg2, const float* __restrict__ bg2,
                            float* __restrict__ mask_out,
                            float* __restrict__ part_o, float* __restrict__ part_ms) {
  extern __shared__ float lds[];
  float* sc = lds;                  // [16 n][LC l][8 h] f32 scaled scores
  float* gw = lds + 16 * LC * 8;    // 81 floats of gate weights

  const int chunk = blockIdx.x;     // 0..63
  const int ntile = blockIdx.y;     // 0..18
  const int b     = blockIdx.z;
  const int h     = threadIdx.x >> 5;   // wave == head
  const int lane  = threadIdx.x & 31;
  const int m  = lane & 15;
  const int ko = (lane >> 4) << 3;
  const int l0 = chunk * LC;

  if (threadIdx.x < 81) {
    int t = threadIdx.x; float v;
    if (t < 64)      v = Wg1[t];
    else if (t < 72) v = bg1[t - 64];
    else if (t < 80) v = Wg2[t - 72];
    else             v = bg2[0];
    gw[t] = v;
  }

  // A operand: q tile rows for this head (row-clamped at N)
  int qr = ntile * 16 + m; if (qr >= NN) qr = NN - 1;
  const __bf16* qrow = qp + (((size_t)(b*HH + h) * NN + qr) << 5);
  v16bf Aq = load_ab_bf16(qrow + ko, qrow + 16 + ko);

  // ---- phase 1: raw scaled scores -> LDS ----
  const __bf16* kbase = kp + ((size_t)(b*HH + h) * LL << 5);
  for (int st = 0; st < LC / 16; ++st) {
    int lb = l0 + st * 16;
    const __bf16* krow = kbase + ((size_t)(lb + m) << 5);
    __builtin_prefetch((const void*)(krow + 16 * 32), 0, 0);   // global_prefetch_b8
    v16bf Bv = load_ab_bf16(krow + ko, krow + 16 + ko);
    v8f Sv = wmma_bf16(Aq, Bv, zero8());
    int lcol = lb + m;
    bool masked = kpm[(size_t)b * LL + lcol] != 0;
    int lloc = st * 16 + m;
#pragma unroll
    for (int i = 0; i < 8; ++i) {
      float v = Sv[i] * SCALE;
      if (masked) v = -__builtin_inff();
      int mr = i + ((lane >> 4) << 3);
      sc[((size_t)mr * LC + lloc) * 8 + h] = v;
    }
  }
  __syncthreads();

  // ---- phase 2: gate MLP over heads, stream mask to output ----
  for (int p = threadIdx.x; p < 16 * LC; p += 256) {
    int n = p / LC, l = p % LC;
    int nglob = ntile * 16 + n;
    float sv[8];
#pragma unroll
    for (int i = 0; i < 8; ++i) sv[i] = sc[(size_t)p * 8 + i];
    float g = gw[80];
#pragma unroll
    for (int j = 0; j < 8; ++j) {
      float a = gw[64 + j];
#pragma unroll
      for (int i = 0; i < 8; ++i) a += sv[i] * gw[i * 8 + j];
      a = fmaxf(a, 0.f);
      g += a * gw[72 + j];
    }
    float mv = 1.f / (1.f + __expf(-g));
    if (nglob < NN)
      mask_out[((size_t)(b * NN + nglob)) * LL + l0 + l] = mv;
  }

  // ---- phase 3: per-head online-softmax partials + P@V (WMMA) ----
  float rmax = -__builtin_inff();
  if (lane < 16) {
    for (int l = 0; l < LC; ++l)
      rmax = fmaxf(rmax, sc[((size_t)lane * LC + l) * 8 + h]);
  }
  float myrmax = __shfl(rmax, m, 32);

  const __bf16* vbase = vT + ((size_t)(b*HH + h) * DH) * LL;
  v8f o0 = zero8(), o1 = zero8();
  float psum = 0.f;

  for (int st = 0; st < LC / 32; ++st) {
    int lb = st * 32;
    V16BF P;
#pragma unroll
    for (int i = 0; i < 8; ++i) {
      float e0 = __expf(sc[((size_t)m * LC + lb + ko + i) * 8 + h] - myrmax);
      float e1 = __expf(sc[((size_t)m * LC + lb + 16 + ko + i) * 8 + h] - myrmax);
      psum += e0 + e1;
      P.h[i]     = (__bf16)e0;
      P.h[8 + i] = (__bf16)e1;
    }
    const __bf16* vrow0 = vbase + (size_t)m * LL + l0 + lb;
    o0 = wmma_bf16(P.v, load_ab_bf16(vrow0 + ko, vrow0 + 16 + ko), o0);
    const __bf16* vrow1 = vbase + (size_t)(16 + m) * LL + l0 + lb;
    o1 = wmma_bf16(P.v, load_ab_bf16(vrow1 + ko, vrow1 + 16 + ko), o1);
  }
  float rsum = psum + __shfl_xor(psum, 16, 32);

  size_t pbase = (((size_t)(b*HH + h) * NT + ntile) * NCHUNK + chunk);
  if (lane < 16) {
    part_ms[pbase * 32 + lane * 2 + 0] = rmax;
    part_ms[pbase * 32 + lane * 2 + 1] = rsum;
  }
  float* po = part_o + pbase * 512;
#pragma unroll
  for (int i = 0; i < 8; ++i) {
    int mr = i + ((lane >> 4) << 3);
    po[mr * 32 + m]      = o0[i];
    po[mr * 32 + 16 + m] = o1[i];
  }
}

// ---------------- combine split-K partials ----------------
__global__ void combine_kernel(const float* __restrict__ part_o,
                               const float* __restrict__ part_ms,
                               __bf16* __restrict__ xbf) {
  int ntile = blockIdx.x, h = blockIdx.y, b = blockIdx.z;
  int lane = threadIdx.x;  // 32 lanes = dh
  size_t base = ((size_t)(b*HH + h) * NT + ntile) * NCHUNK;
  float m[16], s[16], o[16];
#pragma unroll
  for (int r = 0; r < 16; ++r) { m[r] = -__builtin_inff(); s[r] = 0.f; o[r] = 0.f; }
  for (int c = 0; c < NCHUNK; ++c) {
    const float* ms = part_ms + (base + c) * 32;
    const float* po = part_o + (base + c) * 512;
#pragma unroll
    for (int r = 0; r < 16; ++r) {
      float mc = ms[r * 2], sc2 = ms[r * 2 + 1];
      float oc = po[r * 32 + lane];
      float nm = fmaxf(m[r], mc);
      float e0 = __expf(m[r] - nm), e1 = __expf(mc - nm);
      o[r] = o[r] * e0 + oc * e1;
      s[r] = s[r] * e0 + sc2 * e1;
      m[r] = nm;
    }
  }
#pragma unroll
  for (int r = 0; r < 16; ++r) {
    int n = ntile * 16 + r;
    if (n < NN)
      xbf[((size_t)(b * NN + n)) * CC + h * DH + lane] = (__bf16)(o[r] / s[r]);
  }
}

// ---------------- x@Wp + bp + residual + LN1 ----------------
__global__ void post1_kernel(const __bf16* __restrict__ xbf, const __bf16* __restrict__ WpT,
                             const float* __restrict__ bp, const float* __restrict__ query,
                             const float* __restrict__ g1, const float* __restrict__ b1n,
                             float* __restrict__ out1f, __bf16* __restrict__ out1bf) {
  __shared__ float xp[16 * CC];
  __shared__ float red[2][16][16];
  __shared__ float mv[16][2];
  const int r0 = blockIdx.x * 16;
  const int wave = threadIdx.x >> 5, lane = threadIdx.x & 31;
  const int m = lane & 15, ko = (lane >> 4) << 3;
  int rr = r0 + m; if (rr >= BB*NN) rr = BB*NN - 1;
  const __bf16* arow = xbf + (size_t)rr * CC;
  for (int s = 0; s < 2; ++s) {
    int ncol0 = (wave * 2 + s) * 16;
    v8f acc = zero8();
    for (int kk = 0; kk < CC; kk += 32) {
      v16bf A = load_ab_bf16(arow + kk + ko, arow + kk + 16 + ko);
      const __bf16* wrow = WpT + (size_t)(ncol0 + m) * CC + kk;
      acc = wmma_bf16(A, load_ab_bf16(wrow + ko, wrow + 16 + ko), acc);
    }
#pragma unroll
    for (int i = 0; i < 8; ++i) {
      int mr = i + ((lane >> 4) << 3);
      xp[mr * CC + ncol0 + m] = acc[i];
    }
  }
  __syncthreads();
  const int tr = threadIdx.x >> 4, tc = threadIdx.x & 15;
  int grow = r0 + tr; bool valid = grow < BB*NN;
  float vals[16]; float s1 = 0.f, s2 = 0.f;
#pragma unroll
  for (int i = 0; i < 16; ++i) {
    int c = tc * 16 + i;
    float v = xp[tr * CC + c] + bp[c] + (valid ? query[(size_t)grow * CC + c] : 0.f);
    vals[i] = v; s1 += v; s2 += v * v;
  }
  red[0][tr][tc] = s1; red[1][tr][tc] = s2;
  __syncthreads();
  if (tc == 0) {
    float a = 0.f, b2 = 0.f;
#pragma unroll
    for (int i = 0; i < 16; ++i) { a += red[0][tr][i]; b2 += red[1][tr][i]; }
    float mean = a * (1.f / CC);
    float var  = b2 * (1.f / CC) - mean * mean;
    mv[tr][0] = mean; mv[tr][1] = rsqrtf(var + 1e-5f);
  }
  __syncthreads();
  float mean = mv[tr][0], inv = mv[tr][1];
#pragma unroll
  for (int i = 0; i < 16; ++i) {
    int c = tc * 16 + i;
    float y = (vals[i] - mean) * inv * g1[c] + b1n[c];
    if (valid) {
      out1f [(size_t)grow * CC + c] = y;
      out1bf[(size_t)grow * CC + c] = (__bf16)y;
    }
  }
}

// ---------------- gelu(out1@W1 + b1) ----------------
__global__ void ffn1_kernel(const __bf16* __restrict__ out1bf, const __bf16* __restrict__ W1T,
                            const float* __restrict__ b1, __bf16* __restrict__ h1) {
  const int r0 = blockIdx.x * 16;
  const int wave = threadIdx.x >> 5, lane = threadIdx.x & 31;
  const int m = lane & 15, ko = (lane >> 4) << 3;
  int rr = r0 + m; if (rr >= BB*NN) rr = BB*NN - 1;
  const __bf16* arow = out1bf + (size_t)rr * CC;
  for (int s = 0; s < 8; ++s) {
    int ncol0 = (wave * 8 + s) * 16;
    v8f acc = zero8();
    for (int kk = 0; kk < CC; kk += 32) {
      v16bf A = load_ab_bf16(arow + kk + ko, arow + kk + 16 + ko);
      const __bf16* wrow = W1T + (size_t)(ncol0 + m) * CC + kk;
      acc = wmma_bf16(A, load_ab_bf16(wrow + ko, wrow + 16 + ko), acc);
    }
    int c = ncol0 + m;
#pragma unroll
    for (int i = 0; i < 8; ++i) {
      int mr = i + ((lane >> 4) << 3);
      int r = r0 + mr;
      if (r >= BB*NN) continue;
      float x = acc[i] + b1[c];
      float g = 0.5f * x * (1.f + erff(x * 0.70710678118f));
      h1[(size_t)r * HID + c] = (__bf16)g;
    }
  }
}

// ---------------- h1@W2 + b2 + residual + LN2 -> final out ----------------
__global__ void ffn2_kernel(const __bf16* __restrict__ h1, const __bf16* __restrict__ W2T,
                            const float* __restrict__ b2, const float* __restrict__ out1f,
                            const float* __restrict__ g2, const float* __restrict__ b2n,
                            float* __restrict__ outp) {
  __shared__ float xp[16 * CC];
  __shared__ float red[2][16][16];
  __shared__ float mv[16][2];
  const int r0 = blockIdx.x * 16;
  const int wave = threadIdx.x >> 5, lane = threadIdx.x & 31;
  const int m = lane & 15, ko = (lane >> 4) << 3;
  int rr = r0 + m; if (rr >= BB*NN) rr = BB*NN - 1;
  const __bf16* arow = h1 + (size_t)rr * HID;
  for (int s = 0; s < 2; ++s) {
    int ncol0 = (wave * 2 + s) * 16;
    v8f acc = zero8();
    for (int kk = 0; kk < HID; kk += 32) {
      v16bf A = load_ab_bf16(arow + kk + ko, arow + kk + 16 + ko);
      const __bf16* wrow = W2T + (size_t)(ncol0 + m) * HID + kk;
      acc = wmma_bf16(A, load_ab_bf16(wrow + ko, wrow + 16 + ko), acc);
    }
#pragma unroll
    for (int i = 0; i < 8; ++i) {
      int mr = i + ((lane >> 4) << 3);
      xp[mr * CC + ncol0 + m] = acc[i];
    }
  }
  __syncthreads();
  const int tr = threadIdx.x >> 4, tc = threadIdx.x & 15;
  int grow = r0 + tr; bool valid = grow < BB*NN;
  float vals[16]; float s1 = 0.f, s2 = 0.f;
#pragma unroll
  for (int i = 0; i < 16; ++i) {
    int c = tc * 16 + i;
    float v = xp[tr * CC + c] + b2[c] + (valid ? out1f[(size_t)grow * CC + c] : 0.f);
    vals[i] = v; s1 += v; s2 += v * v;
  }
  red[0][tr][tc] = s1; red[1][tr][tc] = s2;
  __syncthreads();
  if (tc == 0) {
    float a = 0.f, bq = 0.f;
#pragma unroll
    for (int i = 0; i < 16; ++i) { a += red[0][tr][i]; bq += red[1][tr][i]; }
    float mean = a * (1.f / CC);
    float var  = bq * (1.f / CC) - mean * mean;
    mv[tr][0] = mean; mv[tr][1] = rsqrtf(var + 1e-5f);
  }
  __syncthreads();
  float mean = mv[tr][0], inv = mv[tr][1];
#pragma unroll
  for (int i = 0; i < 16; ++i) {
    int c = tc * 16 + i;
    float y = (vals[i] - mean) * inv * g2[c] + b2n[c];
    if (valid) outp[(size_t)grow * CC + c] = y;
  }
}

// ---------------- host launcher ----------------
extern "C" void kernel_launch(void* const* d_in, const int* in_sizes, int n_in,
                              void* d_out, int out_size, void* d_ws, size_t ws_size,
                              hipStream_t stream) {
  (void)in_sizes; (void)n_in; (void)out_size; (void)ws_size;
  const float* query = (const float*)d_in[0];
  const float* key   = (const float*)d_in[1];
  const float* value = (const float*)d_in[2];
  const unsigned char* kpm = (const unsigned char*)d_in[3];
  const float* Wq = (const float*)d_in[4];
  const float* Wk = (const float*)d_in[5];
  const float* Wv = (const float*)d_in[6];
  const float* Wp = (const float*)d_in[7];
  const float* bp = (const float*)d_in[8];
  const float* Wg1 = (const float*)d_in[9];
  const float* bg1 = (const float*)d_in[10];
  const float* Wg2 = (const float*)d_in[11];
  const float* bg2 = (const float*)d_in[12];
  const float* ln1g = (const float*)d_in[13];
  const float* ln1b = (const float*)d_in[14];
  const float* W1 = (const float*)d_in[15];
  const float* b1 = (const float*)d_in[16];
  const float* W2 = (const float*)d_in[17];
  const float* b2 = (const float*)d_in[18];
  const float* ln2g = (const float*)d_in[19];
  const float* ln2b = (const float*)d_in[20];

  char* ws = (char*)d_ws;
  size_t off = 0;
  auto alloc = [&](size_t bytes) -> void* {
    void* p = ws + off;
    off += (bytes + 255) & ~(size_t)255;
    return p;
  };
  __bf16* WqT = (__bf16*)alloc((size_t)CC * CC * 2);
  __bf16* WkT = (__bf16*)alloc((size_t)CC * CC * 2);
  __bf16* WvT = (__bf16*)alloc((size_t)CC * CC * 2);
  __bf16* WpT = (__bf16*)alloc((size_t)CC * CC * 2);
  __bf16* W1T = (__bf16*)alloc((size_t)CC * HID * 2);
  __bf16* W2T = (__bf16*)alloc((size_t)HID * CC * 2);
  __bf16* qp  = (__bf16*)alloc((size_t)BB * HH * NN * DH * 2);
  __bf16* kp  = (__bf16*)alloc((size_t)BB * HH * LL * DH * 2);
  __bf16* vT  = (__bf16*)alloc((size_t)BB * HH * DH * LL * 2);
  float* part_o  = (float*)alloc((size_t)BB * HH * NT * NCHUNK * 512 * 4);
  float* part_ms = (float*)alloc((size_t)BB * HH * NT * NCHUNK * 32 * 4);
  __bf16* xbf   = (__bf16*)alloc((size_t)BB * NN * CC * 2);
  float* out1f  = (float*)alloc((size_t)BB * NN * CC * 4);
  __bf16* out1bf = (__bf16*)alloc((size_t)BB * NN * CC * 2);
  __bf16* h1    = (__bf16*)alloc((size_t)BB * NN * HID * 2);

  // 1) weight convert+transpose to bf16 (out-ch major)
  tconv_kernel<<<(CC*CC + 255)/256, 256, 0, stream>>>(Wq, WqT, CC, CC);
  tconv_kernel<<<(CC*CC + 255)/256, 256, 0, stream>>>(Wk, WkT, CC, CC);
  tconv_kernel<<<(CC*CC + 255)/256, 256, 0, stream>>>(Wv, WvT, CC, CC);
  tconv_kernel<<<(CC*CC + 255)/256, 256, 0, stream>>>(Wp, WpT, CC, CC);
  tconv_kernel<<<(CC*HID + 255)/256, 256, 0, stream>>>(W1, W1T, CC, HID);
  tconv_kernel<<<(HID*CC + 255)/256, 256, 0, stream>>>(W2, W2T, HID, CC);

  // 2) q/k/v projections (WMMA bf16)
  proj_kernel<<<38 + 2048 + 2048, 128, 0, stream>>>(query, key, value,
                                                    WqT, WkT, WvT, qp, kp, vT);

  // 3) flash split-K attention + gate-MLP mask (128.3KB dynamic LDS, WGP has 320KB)
  float* mask_out = (float*)d_out + (size_t)BB * NN * CC;
  const int attn_lds = (16 * LC * 8 + 96) * 4;
  hipFuncSetAttribute((const void*)attn_kernel,
                      hipFuncAttributeMaxDynamicSharedMemorySize, attn_lds);
  attn_kernel<<<dim3(NCHUNK, NT, BB), 256, attn_lds, stream>>>(
      qp, kp, vT, kpm, Wg1, bg1, Wg2, bg2, mask_out, part_o, part_ms);

  // 4) combine partials -> attention output (bf16)
  combine_kernel<<<dim3(NT, HH, BB), 32, 0, stream>>>(part_o, part_ms, xbf);

  // 5) projection + residual + LN1
  post1_kernel<<<RT, 256, 0, stream>>>(xbf, WpT, bp, query, ln1g, ln1b, out1f, out1bf);

  // 6) FFN up + gelu
  ffn1_kernel<<<RT, 256, 0, stream>>>(out1bf, W1T, b1, h1);

  // 7) FFN down + residual + LN2 -> d_out
  ffn2_kernel<<<RT, 256, 0, stream>>>(h1, W2T, b2, out1f, ln2g, ln2b, (float*)d_out);
}
